// GCN_24756191494784
// MI455X (gfx1250) — matrix operand order repaired and checked by reading, not compiled, not run on hardware
//
#include <hip/hip_runtime.h>
#include <hip/hip_bf16.h>

typedef __attribute__((ext_vector_type(16))) _Float16 v16h;
typedef __attribute__((ext_vector_type(8)))  _Float16 v8h;
typedef __attribute__((ext_vector_type(8)))  float    v8f;

// ---------------------------------------------------------------------------
// Degree count: deg[dst]++ over real edges (self loop accounted as +1 later)
// ---------------------------------------------------------------------------
__global__ void gcn_degree_kernel(const int* __restrict__ dst,
                                  int* __restrict__ degcnt, int E) {
    int i = blockIdx.x * blockDim.x + threadIdx.x;
    if (i < E) atomicAdd(&degcnt[dst[i]], 1);
}

// dinv[i] = rsqrt(indegree + 1) ; always > 0 so no zero guard needed
__global__ void gcn_dinv_kernel(const int* __restrict__ degcnt,
                                float* __restrict__ dinv, int n) {
    int i = blockIdx.x * blockDim.x + threadIdx.x;
    if (i < n) dinv[i] = rsqrtf((float)degcnt[i] + 1.0f);
}

// ---------------------------------------------------------------------------
// x: f32 -> f16 with zero padding of tail rows
// ---------------------------------------------------------------------------
__global__ void gcn_cvt_pad_kernel(const float* __restrict__ a,
                                   _Float16* __restrict__ b,
                                   long long nvalid, long long ntotal) {
    long long i = (long long)blockIdx.x * blockDim.x + threadIdx.x;
    if (i < ntotal) b[i] = (i < nvalid) ? (_Float16)a[i] : (_Float16)0.0f;
}

// ---------------------------------------------------------------------------
// Pack W (f32 row-major [K, Nout]) into WMMA B-fragment order (f16).
// In this packing (contiguous k, same as the A loads), half index i of lane L holds
//   K = c*32 + (L/16)*16 + i,  col = t*16 + (L%16)
// i.e. each lane holds 16 consecutive K values of one column. Stored so the
// GEMM reads each fragment as two contiguous, 32B-aligned b128 loads.
//   Bp[((t*CH + c)*32 + lane)*16 + i]
// ---------------------------------------------------------------------------
__global__ void gcn_packB_kernel(const float* __restrict__ W,
                                 _Float16* __restrict__ Bp,
                                 int Nout, int chunks, int total) {
    int idx = blockIdx.x * blockDim.x + threadIdx.x;
    if (idx >= total) return;
    int i    = idx & 15;
    int lane = (idx >> 4) & 31;
    int c    = (idx >> 9) % chunks;
    int t    = idx / (512 * chunks);
    int k    = c * 32 + (lane >> 4) * 16 + i;
    int ncol = t * 16 + (lane & 15);
    Bp[idx] = (_Float16)W[k * Nout + ncol];
}

// ---------------------------------------------------------------------------
// WMMA GEMM: C[M, NTILES*16] = A[Mpad, KD] (f16) x Bp (packed f16 fragments).
// One wave32 per 16-row stripe; the wave keeps NTILES accumulators and reuses
// each A fragment across all N tiles. K loop fully unrolled (KD=96 -> 3 WMMA
// chunks, NTILES WMMAs each).
// ---------------------------------------------------------------------------
template <int KD, int NTILES>
__global__ __launch_bounds__(128) void gcn_gemm_wmma_kernel(
    const _Float16* __restrict__ A, const _Float16* __restrict__ Bp,
    float* __restrict__ C, int Nrows, int Mtiles)
{
    constexpr int CH   = KD / 32;
    constexpr int Nout = NTILES * 16;

    const int wave  = threadIdx.x >> 5;
    const int lane  = threadIdx.x & 31;
    const int mtile = blockIdx.x * 4 + wave;
    if (mtile >= Mtiles) return;

    const int g    = lane >> 4;      // lane group (0: lanes 0-15, 1: 16-31)
    const int l15  = lane & 15;
    const int arow = mtile * 16 + l15;

    v8f acc[NTILES] = {};

    const _Float16* ap = A + (size_t)arow * KD + g * 16;
    const _Float16* bp = Bp + (size_t)lane * 16;

    #pragma unroll
    for (int c = 0; c < CH; ++c) {
        // A fragment: halves 0-15 = K c*32+g*16 .. +15
        union { v16h v; v8h h[2]; } a;
        a.h[0] = *(const v8h*)(ap + c * 32);
        a.h[1] = *(const v8h*)(ap + c * 32 + 8);

        #pragma unroll
        for (int t = 0; t < NTILES; ++t) {
            union { v16h v; v8h h[2]; } b;
            const _Float16* q = bp + (size_t)(t * CH + c) * 512;
            b.h[0] = *(const v8h*)(q);
            b.h[1] = *(const v8h*)(q + 8);
            acc[t] = __builtin_amdgcn_wmma_f32_16x16x32_f16(
                /*neg_a=*/false, a.v, /*neg_b=*/false, b.v,
                /*c_mod=*/(short)0, acc[t], /*reuse_a=*/false, /*reuse_b=*/false);
        }
    }

    #pragma unroll
    for (int t = 0; t < NTILES; ++t) {
        #pragma unroll
        for (int r = 0; r < 8; ++r) {
            int row = mtile * 16 + r + 8 * g;
            if (row < Nrows) C[(size_t)row * Nout + t * 16 + l15] = acc[t][r];
        }
    }
}

// ---------------------------------------------------------------------------
// Edge scatter: agg[dst,f] += h[src,f] * dinv[src]*dinv[dst]   (f32 atomics)
// ---------------------------------------------------------------------------
template <int F>
__global__ void gcn_scatter_kernel(const int* __restrict__ src,
                                   const int* __restrict__ dst,
                                   const float* __restrict__ h,
                                   const float* __restrict__ dinv,
                                   float* __restrict__ agg, long long total) {
    long long idx    = (long long)blockIdx.x * blockDim.x + threadIdx.x;
    long long stride = (long long)gridDim.x * blockDim.x;
    for (; idx < total; idx += stride) {
        int e = (int)(idx / F);
        int f = (int)(idx % F);
        int s = src[e];
        int d = dst[e];
        float w = dinv[s] * dinv[d];
        atomicAdd(&agg[(size_t)d * F + f], h[(size_t)s * F + f] * w);
    }
}

// ---------------------------------------------------------------------------
// Finalize layer 1: self-loop term + bias, ReLU; emit f32 output and f16 copy
// for layer-2 GEMM (pad rows zeroed).
// ---------------------------------------------------------------------------
__global__ void gcn_finalize1_kernel(const float* __restrict__ agg,
                                     const float* __restrict__ h,
                                     const float* __restrict__ dinv,
                                     const float* __restrict__ bias,
                                     float* __restrict__ out,
                                     _Float16* __restrict__ x1h,
                                     int nrows, long long total /* = padrows*96 */) {
    long long idx = (long long)blockIdx.x * blockDim.x + threadIdx.x;
    if (idx >= total) return;
    int row = (int)(idx / 96);
    int f   = (int)(idx % 96);
    if (row < nrows) {
        float di = dinv[row];
        float v  = agg[idx] + h[idx] * di * di + bias[f];
        v = v > 0.0f ? v : 0.0f;
        out[idx] = v;
        x1h[idx] = (_Float16)v;
    } else {
        x1h[idx] = (_Float16)0.0f;
    }
}

// Finalize layer 2: self-loop term + bias.
__global__ void gcn_finalize2_kernel(const float* __restrict__ agg,
                                     const float* __restrict__ h,
                                     const float* __restrict__ dinv,
                                     const float* __restrict__ bias,
                                     float* __restrict__ out,
                                     long long total /* = n*32 */) {
    long long idx = (long long)blockIdx.x * blockDim.x + threadIdx.x;
    if (idx >= total) return;
    int row = (int)(idx / 32);
    int f   = (int)(idx % 32);
    float di = dinv[row];
    out[idx] = agg[idx] + h[idx] * di * di + bias[f];
}

// ---------------------------------------------------------------------------
// Host launcher
// ---------------------------------------------------------------------------
extern "C" void kernel_launch(void* const* d_in, const int* in_sizes, int n_in,
                              void* d_out, int out_size, void* d_ws, size_t ws_size,
                              hipStream_t stream) {
    (void)n_in; (void)out_size; (void)ws_size;

    constexpr int FIN  = 96;
    constexpr int FHID = 96;
    constexpr int FOUT = 32;
    constexpr int CH   = FIN / 32;       // 3 K-chunks (same K for both layers)

    const float* x  = (const float*)d_in[0];
    const int*   ei = (const int*)d_in[1];
    const float* W1 = (const float*)d_in[2];
    const float* b1 = (const float*)d_in[3];
    const float* W2 = (const float*)d_in[4];
    const float* b2 = (const float*)d_in[5];

    const int n = in_sizes[0] / FIN;     // 50000
    const int E = in_sizes[1] / 2;       // 800000
    const int* src = ei;
    const int* dst = ei + E;

    const int Mpad   = (n + 15) & ~15;   // 50000 is already a multiple of 16
    const int Mtiles = Mpad / 16;

    // Workspace layout (256B aligned). h/agg/xh buffers reused across layers.
    char*  ws  = (char*)d_ws;
    size_t off = 0;
    auto carve = [&](size_t bytes) -> char* {
        char* p = ws + off;
        off = (off + bytes + 255) & ~(size_t)255;
        return p;
    };
    int*      degcnt = (int*)     carve((size_t)n * sizeof(int));
    float*    dinv   = (float*)   carve((size_t)n * sizeof(float));
    _Float16* Xh     = (_Float16*)carve((size_t)Mpad * FIN * sizeof(_Float16)); // also X1h
    _Float16* B1p    = (_Float16*)carve((size_t)(FHID / 16) * CH * 512 * sizeof(_Float16));
    _Float16* B2p    = (_Float16*)carve((size_t)(FOUT / 16) * CH * 512 * sizeof(_Float16));
    float*    hbuf   = (float*)   carve((size_t)n * FHID * sizeof(float));      // h1, then h2
    float*    agg    = (float*)   carve((size_t)n * FHID * sizeof(float));      // agg1, then agg2

    float* out1 = (float*)d_out;                       // [n, 96] relu'd hidden
    float* out2 = (float*)d_out + (size_t)n * FHID;    // [n, 32] final

    const int T = 256;
    auto blocks = [&](long long total) { return (unsigned)((total + T - 1) / T); };

    // 0) zero degree counters + layer-1 aggregation buffer
    hipMemsetAsync(degcnt, 0, (size_t)n * sizeof(int), stream);
    hipMemsetAsync(agg, 0, (size_t)n * FHID * sizeof(float), stream);

    // 1) degree + dinv
    gcn_degree_kernel<<<blocks(E), T, 0, stream>>>(dst, degcnt, E);
    gcn_dinv_kernel<<<blocks(n), T, 0, stream>>>(degcnt, dinv, n);

    // 2) converts: x -> f16, W1/W2 -> packed WMMA B fragments
    gcn_cvt_pad_kernel<<<blocks((long long)Mpad * FIN), T, 0, stream>>>(
        x, Xh, (long long)n * FIN, (long long)Mpad * FIN);
    gcn_packB_kernel<<<blocks((FHID / 16) * CH * 512), T, 0, stream>>>(
        W1, B1p, FHID, CH, (FHID / 16) * CH * 512);
    gcn_packB_kernel<<<blocks((FOUT / 16) * CH * 512), T, 0, stream>>>(
        W2, B2p, FOUT, CH, (FOUT / 16) * CH * 512);

    // 3) GEMM1: h1 = Xh @ W1  (wave32 WMMA, 6 N-tiles per wave, 18 WMMAs)
    gcn_gemm_wmma_kernel<96, 6><<<dim3((Mtiles + 3) / 4), 128, 0, stream>>>(
        Xh, B1p, hbuf, n, Mtiles);

    // 4) edge scatter layer 1
    gcn_scatter_kernel<FHID><<<blocks((long long)E * FHID), T, 0, stream>>>(
        src, dst, hbuf, dinv, agg, (long long)E * FHID);

    // 5) finalize layer 1: self-loop + bias + ReLU -> out1 (f32) and Xh (f16)
    gcn_finalize1_kernel<<<blocks((long long)Mpad * FHID), T, 0, stream>>>(
        agg, hbuf, dinv, b1, out1, Xh, n, (long long)Mpad * FHID);

    // 6) re-zero aggregation buffer for layer 2 (reused)
    hipMemsetAsync(agg, 0, (size_t)n * FOUT * sizeof(float), stream);

    // 7) GEMM2: h2 = X1h @ W2  (2 N-tiles per wave, 6 WMMAs)
    gcn_gemm_wmma_kernel<96, 2><<<dim3((Mtiles + 3) / 4), 128, 0, stream>>>(
        Xh, B2p, hbuf, n, Mtiles);

    // 8) edge scatter layer 2
    gcn_scatter_kernel<FOUT><<<blocks((long long)E * FOUT), T, 0, stream>>>(
        src, dst, hbuf, dinv, agg, (long long)E * FOUT);

    // 9) finalize layer 2: self-loop + bias -> out2
    gcn_finalize2_kernel<<<blocks((long long)n * FOUT), T, 0, stream>>>(
        agg, hbuf, dinv, b2, out2, (long long)n * FOUT);
}
